// SelfAttention_10230612099358
// MI455X (gfx1250) — matrix operand order repaired and checked by reading, not compiled, not run on hardware
//
#include <hip/hip_runtime.h>

// ---------------------------------------------------------------------------
// Self-attention block for MI455X (gfx1250, wave32, WMMA).
// Compute-bound (~103 GFLOP vs ~50MB traffic -> V_WMMA_F32_16X16X32_F16,
// f16 inputs / f32 accumulate). Round 3: ping-pong K-loops unrolled x2 so
// double-buffer rotation happens by reloading registers (no v_mov chains);
// attention prefetches next block's K/V fragments into the just-freed regs.
// ---------------------------------------------------------------------------

typedef __attribute__((ext_vector_type(16))) _Float16 v16h;
typedef __attribute__((ext_vector_type(8)))  float    v8f;

#define DMODEL 1024
#define TSEQ   2048
#define NH     16
#define DH     64
#define BTOT   2

union Frag16 { v16h v; uint4 u[2]; };

// A-matrix 16x32 f16 fragment (ISA 7.12.2): per lane, two contiguous 8-half
// chunks at K = base+{0..7} and base+{16..23}, base = (lane>>4)*8.
__device__ __forceinline__ v16h frag_a_load(const _Float16* p) {
    Frag16 f;
    f.u[0] = *(const uint4*)(p);
    f.u[1] = *(const uint4*)(p + 16);
    return f.v;
}
// B-matrix 32x16 f16 fragment: per lane two contiguous 8-half chunks at
// K = base+{0..7} and base+{16..23}, base = (lane>>4)*8 (column n = lane&15 fixed).
__device__ __forceinline__ v16h frag_b_load(const _Float16* p) {
    Frag16 f;
    f.u[0] = *(const uint4*)(p);
    f.u[1] = *(const uint4*)(p + 16);
    return f.v;
}
__device__ __forceinline__ v8f wmma_f16(v16h a, v16h b, v8f c) {
    // (neg_a, A, neg_b, B, c_mod, C, reuse_a, reuse_b)
    return __builtin_amdgcn_wmma_f32_16x16x32_f16(false, a, false, b,
                                                  (short)0, c, false, false);
}

// ---------------------------------------------------------------------------
// Kernel 1: cast x (f32 -> f16)
// ---------------------------------------------------------------------------
__global__ void cast_x_f16(const float* __restrict__ x,
                           _Float16* __restrict__ xh, int n) {
    int i = blockIdx.x * blockDim.x + threadIdx.x;
    if (i < n) xh[i] = (_Float16)x[i];
}

// ---------------------------------------------------------------------------
// Kernel 2: transpose+cast weights [K][N] f32 -> [N][K] f16 (LDS 32x32 tile).
// Output packed: Wq^T, Wk^T, Wv^T, Wout^T each 1M halves, consecutive.
// ---------------------------------------------------------------------------
__global__ void transpose_w_f16(const float* __restrict__ wq,
                                const float* __restrict__ wk,
                                const float* __restrict__ wv,
                                const float* __restrict__ wo,
                                _Float16* __restrict__ out) {
    __shared__ float tile[32][33];
    const int mat = blockIdx.z;
    const float* src = (mat == 0) ? wq : (mat == 1) ? wk : (mat == 2) ? wv : wo;
    _Float16* dst = out + (size_t)mat * DMODEL * DMODEL;
    const int bx = blockIdx.x * 32;   // n-dim of src
    const int by = blockIdx.y * 32;   // k-dim of src
    for (int yy = threadIdx.y; yy < 32; yy += 8)
        tile[yy][threadIdx.x] = src[(size_t)(by + yy) * DMODEL + bx + threadIdx.x];
    __syncthreads();
    for (int yy = threadIdx.y; yy < 32; yy += 8)
        dst[(size_t)(bx + yy) * DMODEL + by + threadIdx.x] =
            (_Float16)tile[threadIdx.x][yy];
}

// ---------------------------------------------------------------------------
// Ping-pong GEMM core: 16(M) x 64(N) tile, K = DMODEL, two K-steps per
// iteration; each buffer is refilled by loads (never register moves).
// ---------------------------------------------------------------------------
__device__ __forceinline__ void gemm_16x64_acc(const _Float16* __restrict__ arow,
                                               const _Float16* __restrict__ brow,
                                               v8f acc[4]) {
    v16h a0 = frag_a_load(arow);
    v16h b0[4];
#pragma unroll
    for (int s = 0; s < 4; ++s) b0[s] = frag_b_load(brow + s * 16 * DMODEL);

    for (int kk = 0; kk < DMODEL; kk += 64) {
        // stage 0: issue kk+32 loads, consume kk
        v16h a1 = frag_a_load(arow + kk + 32);
        v16h b1[4];
#pragma unroll
        for (int s = 0; s < 4; ++s)
            b1[s] = frag_b_load(brow + s * 16 * DMODEL + kk + 32);
#pragma unroll
        for (int s = 0; s < 4; ++s) acc[s] = wmma_f16(a0, b0[s], acc[s]);
        // stage 1: refill buf0 from kk+64 (wraps harmlessly on last iter),
        //          consume kk+32
        const int kn = (kk + 64) & (DMODEL - 1);
        a0 = frag_a_load(arow + kn);
#pragma unroll
        for (int s = 0; s < 4; ++s)
            b0[s] = frag_b_load(brow + s * 16 * DMODEL + kn);
#pragma unroll
        for (int s = 0; s < 4; ++s) acc[s] = wmma_f16(a1, b1[s], acc[s]);
    }
}

// ---------------------------------------------------------------------------
// Kernel 3: fused QKV projection + RoPE.
// One wave computes a 16(M) x 64(N) tile = one head-slice; mat in {q,k,v}.
// Q,K get RoPE (even/odd dims are adjacent lanes -> shfl_xor(1)); 1/sqrt(DH)
// folded into Q. Q,K stored [B,H,T,D]; V stored transposed [B,H,D,T].
// ---------------------------------------------------------------------------
__global__ __launch_bounds__(128) void qkv_rope_kernel(
        const _Float16* __restrict__ xh, const _Float16* __restrict__ wT,
        const float* __restrict__ cosp,  const float* __restrict__ sinp,
        _Float16* __restrict__ qh, _Float16* __restrict__ kh,
        _Float16* __restrict__ vT) {
    const int lane   = threadIdx.x & 31;
    const int wid    = threadIdx.x >> 5;
    const int flat   = blockIdx.x * 4 + wid;    // 12288 waves
    const int mat    = flat >> 12;              // 0=Q 1=K 2=V
    const int rem    = flat & 4095;
    const int m0     = (rem >> 4) << 4;         // row tile (b*T+t)
    const int head   = rem & 15;
    const int n0     = head << 6;
    const int lanelo = lane & 15;
    const int lanehi = lane >> 4;

    const _Float16* w = wT + (size_t)mat * DMODEL * DMODEL;
    const _Float16* arow = xh + (size_t)(m0 + lanelo) * DMODEL + (lanehi << 3);
    const _Float16* brow = w + (size_t)(n0 + lanelo) * DMODEL + (lanehi << 3);

    v8f acc[4] = {};
    gemm_16x64_acc(arow, brow, acc);

#pragma unroll
    for (int s = 0; s < 4; ++s) {
#pragma unroll
        for (int r = 0; r < 8; ++r) {
            float vval = acc[s][r];
            const int M = (lanehi << 3) + r;
            const int m = m0 + M;
            const int b = m >> 11;            // T=2048
            const int t = m & (TSEQ - 1);
            const int d = (s << 4) + lanelo;
            const size_t bh = (size_t)(b * NH + head);
            if (mat == 2) {                   // V: transposed [B,H,D,T]
                vT[(bh * DH + d) * TSEQ + t] = (_Float16)vval;
            } else {                          // Q/K: RoPE, layout [B,H,T,D]
                float part = __shfl_xor(vval, 1, 32);
                float c  = cosp[t * (DH / 2) + (d >> 1)];
                float sn = sinp[t * (DH / 2) + (d >> 1)];
                // even lane: x1*c - x2*s ; odd lane: x1*s + x2*c
                float o = (lane & 1) ? (part * sn + vval * c)
                                     : (vval * c - part * sn);
                if (mat == 0)
                    qh[(bh * TSEQ + t) * DH + d] = (_Float16)(o * 0.125f);
                else
                    kh[(bh * TSEQ + t) * DH + d] = (_Float16)o;
            }
        }
    }
}

// ---------------------------------------------------------------------------
// Attention fragment loaders (fully inlined; refill-by-load, no copies)
// ---------------------------------------------------------------------------
__device__ __forceinline__ void load_kfrags(const _Float16* __restrict__ kbase,
                                            int k0, v16h kf[4]) {
    const _Float16* krow = kbase + (size_t)k0 * DH;
    kf[0] = frag_b_load(krow);                 // keys k0..k0+15, d 0..31
    kf[1] = frag_b_load(krow + 32);            // keys k0..k0+15, d 32..63
    kf[2] = frag_b_load(krow + 16 * DH);       // keys k0+16..31, d 0..31
    kf[3] = frag_b_load(krow + 16 * DH + 32);  // keys k0+16..31, d 32..63
}
__device__ __forceinline__ void load_vfrags(const _Float16* __restrict__ vbase,
                                            int k0, v16h vf[4]) {
#pragma unroll
    for (int s = 0; s < 4; ++s)
        vf[s] = frag_b_load(vbase + (size_t)(s * 16) * TSEQ + k0);
}

// ---------------------------------------------------------------------------
// Kernel 4: causal flash attention. One wave per (b,h,16-query tile).
// 32 keys/iter. Software pipeline: K frags for block i+1 are loaded right
// after the S WMMAs of block i (latency hidden by softmax + LDS turnaround);
// V frags for block i+1 are loaded right after P*V of block i (hidden by the
// next block's S + softmax). P is bounced through per-wave LDS
// (C-layout -> A-layout). No block barriers (different trip counts per wave).
// ---------------------------------------------------------------------------
__global__ __launch_bounds__(128) void flash_attn_kernel(
        const _Float16* __restrict__ qh, const _Float16* __restrict__ kh,
        const _Float16* __restrict__ vT, _Float16* __restrict__ zh) {
    __shared__ __align__(16) _Float16 plds[4 * 16 * 32];
    const int lane   = threadIdx.x & 31;
    const int wid    = threadIdx.x >> 5;
    const int flat   = blockIdx.x * 4 + wid;   // 4096 waves
    const int b      = flat >> 11;
    const int h      = (flat >> 7) & 15;
    const int q0     = (flat & 127) << 4;
    const int lanelo = lane & 15;
    const int lanehi = lane >> 4;
    const size_t bh  = (size_t)(b * NH + h);

    const _Float16* qrow = qh + (bh * TSEQ + q0 + lanelo) * DH + (lanehi << 3);
    v16h qa0 = frag_a_load(qrow);        // d 0..31
    v16h qa1 = frag_a_load(qrow + 32);   // d 32..63

    v8f o[4] = {};
    float mrow[8], lrow[8];
#pragma unroll
    for (int r = 0; r < 8; ++r) { mrow[r] = -3.0e38f; lrow[r] = 0.0f; }

    _Float16* pw = plds + wid * 512;
    const _Float16* kbase = kh + (bh * TSEQ + lanelo) * DH + (lanehi << 3);
    const _Float16* vbase = vT + (bh * DH + lanelo) * TSEQ + (lanehi << 3);

    v16h kf[4], vf[4];
    load_kfrags(kbase, 0, kf);
    load_vfrags(vbase, 0, vf);

    const int nkb = ((q0 + 15) >> 5) + 1;
    for (int kbi = 0; kbi < nkb; ++kbi) {
        const int k0 = kbi << 5;
        const int kn = (k0 + 32) & (TSEQ - 1);   // next block, wraps in-bounds

        // ---- scores (kf loaded one block ahead) ----
        v8f S[2] = {};
        S[0] = wmma_f16(qa0, kf[0], S[0]);
        S[0] = wmma_f16(qa1, kf[1], S[0]);
        S[1] = wmma_f16(qa0, kf[2], S[1]);
        S[1] = wmma_f16(qa1, kf[3], S[1]);
        // kf is dead: refill for next block while softmax runs
        load_kfrags(kbase, kn, kf);

        // ---- online softmax ----
        const bool needmask = (k0 + 31 > q0);
#pragma unroll
        for (int r = 0; r < 8; ++r) {
            const int qi = q0 + (lanehi << 3) + r;
            float s0 = S[0][r], s1 = S[1][r];
            if (needmask) {
                if (k0 + lanelo      > qi) s0 = -3.0e38f;
                if (k0 + 16 + lanelo > qi) s1 = -3.0e38f;
            }
            float mx = fmaxf(s0, s1);
#pragma unroll
            for (int off = 1; off < 16; off <<= 1)
                mx = fmaxf(mx, __shfl_xor(mx, off, 32));
            const float mnew  = fmaxf(mrow[r], mx);
            const float alpha = __expf(mrow[r] - mnew);
            const float p0 = __expf(s0 - mnew);
            const float p1 = __expf(s1 - mnew);
            float rs = p0 + p1;
#pragma unroll
            for (int off = 1; off < 16; off <<= 1)
                rs += __shfl_xor(rs, off, 32);
            lrow[r] = lrow[r] * alpha + rs;
            mrow[r] = mnew;
#pragma unroll
            for (int s = 0; s < 4; ++s) o[s][r] *= alpha;
            const int M = (lanehi << 3) + r;
            pw[M * 32 + lanelo]      = (_Float16)p0;
            pw[M * 32 + 16 + lanelo] = (_Float16)p1;
        }
        // LDS turnaround within the wave: make stores visible before re-read
        asm volatile("s_wait_dscnt 0" ::: "memory");
        const _Float16* prow = pw + lanelo * 32 + (lanehi << 3);
        v16h pa = frag_a_load(prow);   // P in A-layout (16q x 32k)

        // ---- P x V (vf loaded one block ahead) ----
#pragma unroll
        for (int s = 0; s < 4; ++s) o[s] = wmma_f16(pa, vf[s], o[s]);
        // vf is dead: refill for next block (hidden by next S + softmax)
        load_vfrags(vbase, kn, vf);
    }
    // epilogue: normalize, store z as f16 [B,T,H*D] for the final GEMM
#pragma unroll
    for (int r = 0; r < 8; ++r) {
        const float inv = 1.0f / lrow[r];
        const int M = (lanehi << 3) + r;
        const size_t row = (size_t)(b * TSEQ + q0 + M);
#pragma unroll
        for (int s = 0; s < 4; ++s)
            zh[row * DMODEL + h * DH + s * 16 + lanelo] =
                (_Float16)(o[s][r] * inv);
    }
}

// ---------------------------------------------------------------------------
// Kernel 5: output projection z(4096x1024 f16) @ Wout -> f32 out.
// ---------------------------------------------------------------------------
__global__ __launch_bounds__(128) void out_proj_kernel(
        const _Float16* __restrict__ zh, const _Float16* __restrict__ woT,
        float* __restrict__ out) {
    const int lane   = threadIdx.x & 31;
    const int wid    = threadIdx.x >> 5;
    const int flat   = blockIdx.x * 4 + wid;   // 4096 waves
    const int m0     = (flat >> 4) << 4;
    const int n0     = (flat & 15) << 6;
    const int lanelo = lane & 15;
    const int lanehi = lane >> 4;

    const _Float16* arow = zh + (size_t)(m0 + lanelo) * DMODEL + (lanehi << 3);
    const _Float16* brow = woT + (size_t)(n0 + lanelo) * DMODEL + (lanehi << 3);

    v8f acc[4] = {};
    gemm_16x64_acc(arow, brow, acc);

#pragma unroll
    for (int s = 0; s < 4; ++s)
#pragma unroll
        for (int r = 0; r < 8; ++r)
            out[(size_t)(m0 + (lanehi << 3) + r) * DMODEL + n0 + s * 16 + lanelo] =
                acc[s][r];
}

// ---------------------------------------------------------------------------
extern "C" void kernel_launch(void* const* d_in, const int* in_sizes, int n_in,
                              void* d_out, int out_size, void* d_ws, size_t ws_size,
                              hipStream_t stream) {
    (void)in_sizes; (void)n_in; (void)out_size; (void)ws_size;
    const float* x    = (const float*)d_in[0];
    const float* cosp = (const float*)d_in[1];
    const float* sinp = (const float*)d_in[2];
    const float* wq   = (const float*)d_in[3];
    const float* wk   = (const float*)d_in[4];
    const float* wv   = (const float*)d_in[5];
    const float* wo   = (const float*)d_in[6];
    float* out = (float*)d_out;

    // Workspace layout (48 MB total):
    //  [ 0, 8)MB  xh   f16 [4096][1024]
    //  [ 8,14)MB  wT   f16 WqT,WkT,WvT packed [N][K]
    //  [14,16)MB  woT  f16 WoutT [N][K]
    //  [16,24)MB  qh   f16 [B,H,T,D] (x0.125 folded in)
    //  [24,32)MB  kh   f16 [B,H,T,D]
    //  [32,40)MB  vT   f16 [B,H,D,T]
    //  [40,48)MB  zh   f16 [B,T,H*D]
    char* ws = (char*)d_ws;
    const size_t MB = 1024 * 1024;
    _Float16* xh  = (_Float16*)(ws);
    _Float16* wT  = (_Float16*)(ws + 8 * MB);
    _Float16* woT = (_Float16*)(ws + 14 * MB);
    _Float16* qh  = (_Float16*)(ws + 16 * MB);
    _Float16* kh  = (_Float16*)(ws + 24 * MB);
    _Float16* vT  = (_Float16*)(ws + 32 * MB);
    _Float16* zh  = (_Float16*)(ws + 40 * MB);

    const int nx = BTOT * TSEQ * DMODEL;                 // 4,194,304
    cast_x_f16<<<(nx + 255) / 256, 256, 0, stream>>>(x, xh, nx);
    transpose_w_f16<<<dim3(32, 32, 4), dim3(32, 8), 0, stream>>>(wq, wk, wv, wo, wT);
    qkv_rope_kernel<<<3072, 128, 0, stream>>>(xh, wT, cosp, sinp, qh, kh, vT);
    flash_attn_kernel<<<1024, 128, 0, stream>>>(qh, kh, vT, zh);
    out_proj_kernel<<<1024, 128, 0, stream>>>(zh, woT, out);
}
